// ParallelMHA_84662395339193
// MI455X (gfx1250) — matrix operand order repaired and checked by reading, not compiled, not run on hardware
//
#include <hip/hip_runtime.h>
#include <hip/hip_bf16.h>
#include <math.h>

// ---------------------------------------------------------------------------
// ParallelMHA forward for MI455X (gfx1250, wave32, WMMA).
// All GEMM-like contractions use v_wmma_f32_16x16x32_bf16 (bf16 in, f32 acc).
// GEMM waves own 16x64 output tiles (4 accumulators) to amortize A traffic.
// Attention K and V tiles staged to LDS with GLOBAL_LOAD_ASYNC_TO_LDS_B128,
// double-buffered so DMA of tile i+1 overlaps the 16 WMMAs of tile i.
// V is produced pre-transposed globally (vT[B,H,D,S]) so its LDS tile is a
// direct async copy and PV B-fragments read contiguous bf16 from LDS.
// ---------------------------------------------------------------------------

typedef __attribute__((ext_vector_type(16))) __bf16 bf16x16;
typedef __attribute__((ext_vector_type(8)))  float  f32x8;

#define B_    2
#define S_    2048
#define E_    2048
#define H_    16
#define D_    128
#define QKV_  (3 * E_)      // 6144
#define ROWS_ (B_ * S_)     // 4096
#define VTS_  40            // padded row stride (bf16) of transposed V tile

#if defined(__has_builtin)
#if __has_builtin(__builtin_amdgcn_global_load_async_to_lds_b128) && \
    __has_builtin(__builtin_amdgcn_s_wait_asynccnt)
#define USE_ASYNC_LDS 1
#endif
#endif
#ifndef USE_ASYNC_LDS
#define USE_ASYNC_LDS 0
#endif

typedef int i32x4_t __attribute__((__vector_size__(16)));

__device__ __forceinline__ f32x8 zero8() {
  f32x8 z;
#pragma unroll
  for (int j = 0; j < 8; ++j) z[j] = 0.0f;
  return z;
}

// ---------------------------------------------------------------------------
// fp32 -> bf16 conversion
// ---------------------------------------------------------------------------
__global__ void cvt_f32_bf16(const float* __restrict__ src,
                             __bf16* __restrict__ dst, int n) {
  int i = blockIdx.x * blockDim.x + threadIdx.x;
  if (i < n) dst[i] = (__bf16)src[i];
}

// ---------------------------------------------------------------------------
// NT mainloop, one wave -> 16x64 tile: 4 f32 accumulators share one A frag.
// All four B fragments loaded before the 4 WMMAs for staggered load waits.
// ---------------------------------------------------------------------------
__device__ __forceinline__ void gemm_tile_nt4(const __bf16* __restrict__ A,
                                              const __bf16* __restrict__ Bt,
                                              int K, int r, int hs,
                                              f32x8 acc[4]) {
#pragma unroll
  for (int n = 0; n < 4; ++n) acc[n] = zero8();
  const __bf16* arow = A + (size_t)r * K;
  const __bf16* brow = Bt + (size_t)r * K;
  for (int k0 = 0; k0 < K; k0 += 32) {
    bf16x16 af, bf[4];
    const __bf16* ap = arow + k0 + hs * 16;
#pragma unroll
    for (int j = 0; j < 8; ++j) { af[j] = ap[j]; af[8 + j] = ap[8 + j]; }
#pragma unroll
    for (int n = 0; n < 4; ++n) {
      const __bf16* bp = brow + (size_t)n * 16 * K + k0 + hs * 16;
#pragma unroll
      for (int j = 0; j < 16; ++j) bf[n][j] = bp[j];
      __builtin_prefetch(bp + 64, 0, 1);  // global_prefetch_b8
    }
    __builtin_prefetch(ap + 64, 0, 1);
#pragma unroll
    for (int n = 0; n < 4; ++n)
      acc[n] = __builtin_amdgcn_wmma_f32_16x16x32_bf16(false, af, false, bf[n],
                                                       (short)0, acc[n],
                                                       false, false);
  }
}

// ---------------------------------------------------------------------------
// qkv = x @ Wqkv^T + b ; scatter into q/k as bf16 [B,H,S,D] and v as
// TRANSPOSED bf16 vT[B,H,D,S] (so attention can async-DMA V tiles).
// wave -> 16x64 tile; tiles = 256 x 96; grid = 24576/8 = 3072 blocks
// ---------------------------------------------------------------------------
__global__ __launch_bounds__(256) void gemm_qkv(
    const __bf16* __restrict__ X, const __bf16* __restrict__ W,
    const float* __restrict__ bias, __bf16* __restrict__ qh,
    __bf16* __restrict__ kh, __bf16* __restrict__ vT) {
  const int tiles_n = QKV_ / 64;  // 96
  int wave = threadIdx.x >> 5, lane = threadIdx.x & 31;
  int tile = blockIdx.x * 8 + wave;
  if (tile >= (ROWS_ / 16) * tiles_n) return;
  int tm = tile / tiles_n, tn = tile % tiles_n;
  int r = lane & 15, hs = lane >> 4;

  f32x8 acc[4];
  gemm_tile_nt4(X + (size_t)tm * 16 * E_, W + (size_t)tn * 64 * E_, E_, r, hs,
                acc);

#pragma unroll
  for (int n = 0; n < 4; ++n) {
    int col = tn * 64 + n * 16 + r;       // feature index f in [0,6144)
    float bv = bias[col];
    int three = col >> 11;                // 0:q 1:k 2:v
    int h = (col >> 7) & (H_ - 1);
    int d = col & (D_ - 1);
    __bf16* dst = (three == 0) ? qh : ((three == 1) ? kh : vT);
#pragma unroll
    for (int jj = 0; jj < 8; ++jj) {      // C layout: M = jj + hs*8, N = r
      int row = tm * 16 + jj + hs * 8;
      int b = row >> 11, t = row & (S_ - 1);
      size_t idx = (three == 2)
          ? (((size_t)(b * H_ + h)) * D_ + d) * S_ + t   // vT[b,h,d,t]
          : (((size_t)(b * H_ + h)) * S_ + t) * D_ + d;  // q/k[b,h,t,d]
      dst[idx] = (__bf16)(acc[n][jj] + bv);
    }
  }
}

// ---------------------------------------------------------------------------
// in-place rotary on bf16 [B,H,S,D]; one thread per (b,h,t,i<64) pair
// ---------------------------------------------------------------------------
__global__ void rotary_inplace(__bf16* __restrict__ buf, int n) {
  int idx = blockIdx.x * blockDim.x + threadIdx.x;
  if (idx >= n) return;
  int i = idx & 63;
  int t = (idx >> 6) & (S_ - 1);
  int bh = idx >> 17;
  size_t base = ((size_t)bh * S_ + t) * D_;
  float inv = __expf(-(float)i * 0.14391156831212787f);  // ln(10000)/64
  float ang = (float)t * inv;
  float sn, cs;
  __sincosf(ang, &sn, &cs);
  float x1 = (float)buf[base + i];
  float x2 = (float)buf[base + 64 + i];
  buf[base + i]      = (__bf16)(x1 * cs - x2 * sn);
  buf[base + 64 + i] = (__bf16)(x1 * sn + x2 * cs);
}

// ---------------------------------------------------------------------------
// causal flash attention: grid (B*H, S/128), 256 threads (8 waves).
// Each wave owns 16 Q rows. K tile (row-major) and V tile (transposed,
// padded rows) are staged via async DMA, double-buffered: tile i+1's copies
// are in flight while tile i's 16 WMMAs execute. Async loads complete in
// order per wave, so s_wait_asynccnt<=4 releases the current tile only.
// ---------------------------------------------------------------------------
__global__ __launch_bounds__(256) void attn_fwd(
    const __bf16* __restrict__ qh, const __bf16* __restrict__ kh,
    const __bf16* __restrict__ vT, __bf16* __restrict__ ctx) {
  __shared__ __align__(16) __bf16 Kt[2][32 * 128];      // 16 KB
  __shared__ __align__(16) __bf16 Vt[2][128 * VTS_];    // 20 KB (transposed V)
  __shared__ float  Sw[8][16][32];                      // 16 KB
  __shared__ __bf16 Pw[8][16][32];                      // 8 KB
  __shared__ float  Al[8][16];
  __shared__ float  Ll[8][16];

  int bh = blockIdx.x;            // b*H + h
  int qt0 = blockIdx.y * 128;
  int wave = threadIdx.x >> 5, lane = threadIdx.x & 31;
  int r = lane & 15, hs = lane >> 4;
  int trow = qt0 + wave * 16;
  const int tix = threadIdx.x;

  const __bf16* Q  = qh + (size_t)bh * S_ * D_;
  const __bf16* Kg = kh + (size_t)bh * S_ * D_;
  const __bf16* Vg = vT + (size_t)bh * D_ * S_;

  // stage K tile (linear, 512 x 16B chunks) and V tile (strided rows of the
  // global transposed V into padded LDS rows); 4 async copies per thread.
  auto stage = [&](int buf, int s0) {
#pragma unroll
    for (int part = 0; part < 2; ++part) {
      int c = tix + part * 256;
      // K: chunk c covers bf16 elements [c*8, c*8+8) of the 32x128 tile
      const __bf16* gk = Kg + (size_t)s0 * D_ + (size_t)c * 8;
      __bf16* lk = &Kt[buf][(size_t)c * 8];
      // V: chunk c -> row d = c>>2, column group s8 = (c&3)*8
      int d = c >> 2, s8 = (c & 3) * 8;
      const __bf16* gv = Vg + (size_t)d * S_ + s0 + s8;
      __bf16* lv = &Vt[buf][(size_t)d * VTS_ + s8];
#if USE_ASYNC_LDS
      __builtin_amdgcn_global_load_async_to_lds_b128(
          (__attribute__((address_space(1))) i32x4_t*)gk,
          (__attribute__((address_space(3))) i32x4_t*)lk, 0, 0);
      __builtin_amdgcn_global_load_async_to_lds_b128(
          (__attribute__((address_space(1))) i32x4_t*)gv,
          (__attribute__((address_space(3))) i32x4_t*)lv, 0, 0);
#else
      *(uint4*)lk = *(const uint4*)gk;
      *(uint4*)lv = *(const uint4*)gv;
#endif
    }
  };

  // preload Q fragments for this wave's 16 rows (16 x 128 = 4 A-fragments)
  bf16x16 qf[4];
#pragma unroll
  for (int f = 0; f < 4; ++f) {
    const __bf16* p = Q + (size_t)(trow + r) * D_ + f * 32 + hs * 16;
#pragma unroll
    for (int j = 0; j < 8; ++j) { qf[f][j] = p[j]; qf[f][8 + j] = p[8 + j]; }
  }

  f32x8 o[8];
#pragma unroll
  for (int c = 0; c < 8; ++c) o[c] = zero8();
  float m_i = -1.0e30f, l_i = 0.0f;

  const int nblk = (qt0 + 128) / 32;           // causal bound for this block
  const float scale = 0.088388347648318447f;   // 1/sqrt(128)

  stage(0, 0);                                 // prologue

  for (int it = 0; it < nblk; ++it) {
    int s0 = it * 32;
    int cur = it & 1;
    if (it + 1 < nblk) {
      stage(cur ^ 1, s0 + 32);                 // overlap DMA with compute
#if USE_ASYNC_LDS
      __builtin_amdgcn_s_wait_asynccnt(4);     // current tile's 4 done
#endif
    } else {
#if USE_ASYNC_LDS
      __builtin_amdgcn_s_wait_asynccnt(0);
#endif
    }
    __syncthreads();                           // all lanes' tile-it data in LDS

    // ---- S(16x32) = Q(16x128) @ K_tile(32x128)^T : 8 WMMAs ----
    f32x8 sc[2];
#pragma unroll
    for (int c = 0; c < 2; ++c) {
      f32x8 a = zero8();
#pragma unroll
      for (int f = 0; f < 4; ++f) {
        bf16x16 bfrag;
        const __bf16* p = &Kt[cur][(size_t)(c * 16 + r) * D_ + f * 32 + hs * 16];
#pragma unroll
        for (int j = 0; j < 16; ++j) bfrag[j] = p[j];
        a = __builtin_amdgcn_wmma_f32_16x16x32_bf16(false, qf[f], false, bfrag,
                                                    (short)0, a, false, false);
      }
      sc[c] = a;
    }

    // ---- scale + causal mask, relayout C->rows in LDS ----
#pragma unroll
    for (int c = 0; c < 2; ++c)
#pragma unroll
      for (int jj = 0; jj < 8; ++jj) {
        int M = jj + hs * 8;
        int t = trow + M;
        int s = s0 + c * 16 + r;
        float v = sc[c][jj] * scale;
        if (s > t) v = -3.0e38f;
        Sw[wave][M][c * 16 + r] = v;
      }

    // ---- online softmax: lane r (<16) owns row r (DS in-order in-wave) ----
    if (hs == 0) {
      float mx = m_i;
#pragma unroll
      for (int k = 0; k < 32; ++k) mx = fmaxf(mx, Sw[wave][r][k]);
      float alpha = __expf(m_i - mx);
      float sum = 0.0f;
#pragma unroll
      for (int k = 0; k < 32; ++k) {
        float p = __expf(Sw[wave][r][k] - mx);
        Pw[wave][r][k] = (__bf16)p;
        sum += p;
      }
      l_i = l_i * alpha + sum;
      m_i = mx;
      Al[wave][r] = alpha;
    }

    // ---- rescale accumulators by alpha(row) ----
#pragma unroll
    for (int jj = 0; jj < 8; ++jj) {
      float a = Al[wave][jj + hs * 8];
#pragma unroll
      for (int c = 0; c < 8; ++c) o[c][jj] *= a;
    }

    // ---- O(16x128) += P(16x32) @ V_tile(32x128) : 8 WMMAs ----
    bf16x16 pf;
    {
      const __bf16* p = &Pw[wave][r][0];
#pragma unroll
      for (int j = 0; j < 8; ++j) {
        pf[j]     = p[hs * 16 + j];
        pf[8 + j] = p[hs * 16 + 8 + j];
      }
    }
#pragma unroll
    for (int c = 0; c < 8; ++c) {
      bf16x16 vfrag;
      const __bf16* p = &Vt[cur][(size_t)(c * 16 + r) * VTS_ + hs * 16];
#pragma unroll
      for (int j = 0; j < 16; ++j) vfrag[j] = p[j];
      o[c] = __builtin_amdgcn_wmma_f32_16x16x32_bf16(false, pf, false, vfrag,
                                                     (short)0, o[c], false, false);
    }
    __syncthreads();   // everyone done reading buf 'cur' before it's restaged
  }

  if (hs == 0) Ll[wave][r] = 1.0f / l_i;
  int b = bh >> 4, h = bh & (H_ - 1);
#pragma unroll
  for (int jj = 0; jj < 8; ++jj) {
    int M = jj + hs * 8;
    float linv = Ll[wave][M];
    int t = trow + M;
#pragma unroll
    for (int c = 0; c < 8; ++c) {
      int d = c * 16 + r;
      ctx[((size_t)b * S_ + t) * E_ + h * D_ + d] = (__bf16)(o[c][jj] * linv);
    }
  }
}

// ---------------------------------------------------------------------------
// out = ctx @ out_w^T + out_b  (fp32 output); wave -> 16x64 tile
// tiles = 256 x 32; grid = 8192/8 = 1024 blocks
// ---------------------------------------------------------------------------
__global__ __launch_bounds__(256) void gemm_out(
    const __bf16* __restrict__ C, const __bf16* __restrict__ W,
    const float* __restrict__ bias, float* __restrict__ out) {
  const int tiles_n = E_ / 64;  // 32
  int wave = threadIdx.x >> 5, lane = threadIdx.x & 31;
  int tile = blockIdx.x * 8 + wave;
  if (tile >= (ROWS_ / 16) * tiles_n) return;
  int tm = tile / tiles_n, tn = tile % tiles_n;
  int r = lane & 15, hs = lane >> 4;

  f32x8 acc[4];
  gemm_tile_nt4(C + (size_t)tm * 16 * E_, W + (size_t)tn * 64 * E_, E_, r, hs,
                acc);
#pragma unroll
  for (int n = 0; n < 4; ++n) {
    int col = tn * 64 + n * 16 + r;
    float bv = bias[col];
#pragma unroll
    for (int jj = 0; jj < 8; ++jj) {
      int row = tm * 16 + jj + hs * 8;
      out[(size_t)row * E_ + col] = acc[n][jj] + bv;
    }
  }
}

// ---------------------------------------------------------------------------
extern "C" void kernel_launch(void* const* d_in, const int* in_sizes, int n_in,
                              void* d_out, int out_size, void* d_ws,
                              size_t ws_size, hipStream_t stream) {
  (void)in_sizes; (void)n_in; (void)out_size; (void)ws_size;
  const float* x      = (const float*)d_in[0];
  const float* Wqkv_w = (const float*)d_in[1];
  const float* Wqkv_b = (const float*)d_in[2];
  const float* out_w  = (const float*)d_in[3];
  const float* out_b  = (const float*)d_in[4];
  float* out = (float*)d_out;

  char* ws = (char*)d_ws;
  size_t off = 0;
  auto alloc = [&](size_t bytes) -> char* {
    char* p = ws + off;
    off += (bytes + 255) & ~(size_t)255;
    return p;
  };
  __bf16* xh  = (__bf16*)alloc((size_t)ROWS_ * E_ * 2);         // 16 MB
  __bf16* wqh = (__bf16*)alloc((size_t)QKV_ * E_ * 2);          // 25 MB
  __bf16* owh = (__bf16*)alloc((size_t)E_ * E_ * 2);            // 8 MB
  __bf16* qh  = (__bf16*)alloc((size_t)B_ * H_ * S_ * D_ * 2);  // 16 MB
  __bf16* kh  = (__bf16*)alloc((size_t)B_ * H_ * S_ * D_ * 2);  // 16 MB
  __bf16* vTb = (__bf16*)alloc((size_t)B_ * H_ * S_ * D_ * 2);  // 16 MB
  __bf16* ch  = (__bf16*)alloc((size_t)ROWS_ * E_ * 2);         // 16 MB

  int n1 = ROWS_ * E_;
  cvt_f32_bf16<<<(n1 + 255) / 256, 256, 0, stream>>>(x, xh, n1);
  int n2 = QKV_ * E_;
  cvt_f32_bf16<<<(n2 + 255) / 256, 256, 0, stream>>>(Wqkv_w, wqh, n2);
  int n3 = E_ * E_;
  cvt_f32_bf16<<<(n3 + 255) / 256, 256, 0, stream>>>(out_w, owh, n3);

  // 256 x 96 tiles of 16x64, 8 waves/block
  gemm_qkv<<<3072, 256, 0, stream>>>(xh, wqh, Wqkv_b, qh, kh, vTb);

  int nr = B_ * H_ * S_ * 64;
  rotary_inplace<<<(nr + 255) / 256, 256, 0, stream>>>(qh, nr);
  rotary_inplace<<<(nr + 255) / 256, 256, 0, stream>>>(kh, nr);

  attn_fwd<<<dim3(B_ * H_, S_ / 128), 256, 0, stream>>>(qh, kh, vTb, ch);

  // 256 x 32 tiles of 16x64, 8 waves/block
  gemm_out<<<1024, 256, 0, stream>>>(ch, owh, out_b, out);
}
